// SynthDecoder_37091337568380
// MI455X (gfx1250) — compile-verified, matrix operand order checked by reading, not run on hardware
//
#include <hip/hip_runtime.h>
#include <hip/hip_bf16.h>
#include <math.h>

// ---------------------------------------------------------------------------
// SynthDecoder (DDSP-style) for gfx1250 / MI455X.
//  - GEMMs (conv3-as-K384, pw1, pw2, l1) on v_wmma_f32_16x16x32_f16.
//  - Weights pre-packed to f16 in fragment order -> A frag = one 32B load.
//  - B fragments staged once per block through LDS (ds_load_b128), shared by
//    all 4 waves (16 WMMAs per staging).
//  - Oscillator bank uses closed-form phase (prefix table over 512 frame
//    intervals) -> no serial cumsum, no 256MB audio-rate intermediates.
// ---------------------------------------------------------------------------

typedef __attribute__((ext_vector_type(16))) _Float16 v16h;
typedef __attribute__((ext_vector_type(8)))  float    v8f;

#define Bc 4
#define Tc 512
#define Mc 256          // D2
#define NS 65536
#define SRF 44100.0f
#define NYQ 22050.0f

__device__ __forceinline__ float sigmoidf_(float x) { return 1.0f / (1.0f + __expf(-x)); }
__device__ __forceinline__ float siluf_(float x)    { return x * sigmoidf_(x); }
__device__ __forceinline__ float bnf_(float x, float g, float b, float m, float v) {
    return (x - m) * rsqrtf(v + 1e-5f) * g + b;
}
__device__ __forceinline__ float expsig_(float x) {
    return 2.0f * __powf(sigmoidf_(x), 2.3025850929940457f) + 1e-7f;   // 2*sig^ln10+1e-7
}
__device__ __forceinline__ int imin_(int a, int b) { return a < b ? a : b; }
__device__ __forceinline__ int imax_(int a, int b) { return a > b ? a : b; }
// ISA 7.12.2 16-bit fragment: lane = row + 16*h ; element e -> K offset
__device__ __forceinline__ int wmma_k_(int e, int h) { return 8 * h + e + ((e >= 8) ? 8 : 0); }
// inverse: K offset (0..31) -> (h, e)
__device__ __forceinline__ int inv_h_(int k) { return (k >> 3) & 1; }
__device__ __forceinline__ int inv_e_(int k) { return (k & 7) + ((k >= 16) ? 8 : 0); }

// ---------------------------------------------------------------------------
// Weight pack: fp32 -> f16 fragment order  Wpk[mt][kch][lane][e]
// MODE 0: conv3, K'=384, k' = r*128+c, src W[m*384 + c*3 + r]
// MODE 1/2: pointwise, src W[m*CK + k]
// MODE 3: l1 (transposed), src W[k*256 + m]
// ---------------------------------------------------------------------------
template <int MODE, int CK>
__global__ void pack_k(const float* __restrict__ W, _Float16* __restrict__ Wpk) {
    const int idx = blockIdx.x * blockDim.x + threadIdx.x;
    if (idx >= Mc * CK) return;
    const int e    = idx & 15;
    const int lane = (idx >> 4) & 31;
    const int rem  = idx >> 9;
    const int kch  = rem % (CK / 32);
    const int mt   = rem / (CK / 32);
    const int m    = mt * 16 + (lane & 15);
    const int k    = kch * 32 + wmma_k_(e, lane >> 4);
    float v;
    if (MODE == 0)      v = W[(size_t)m * 384 + (size_t)(k & 127) * 3 + (k >> 7)];
    else if (MODE == 3) v = W[(size_t)k * 256 + m];
    else                v = W[(size_t)m * CK + k];
    Wpk[idx] = (_Float16)v;
}

// ---------------------------------------------------------------------------
// Channel-GEMM on WMMA:  out[b,m,t] = sum_k A[m,k] * X[b,k,t]  (+epilogue)
// grid: Bc*(Tc/16) blocks of 128 threads (4 waves); wave owns 4 M-tiles.
// B chunk (32k x 16t) staged via LDS once per block per chunk.
// ---------------------------------------------------------------------------
template <int MODE, int CK>
__global__ __launch_bounds__(128) void gemm_k(const float* __restrict__ X,
                                              const _Float16* __restrict__ Wpk,
                                              float* __restrict__ out,
                                              const float* __restrict__ bias,
                                              const float* __restrict__ bng,
                                              const float* __restrict__ bnb,
                                              const float* __restrict__ bnm,
                                              const float* __restrict__ bnv,
                                              const float* __restrict__ resid) {
    __shared__ __attribute__((aligned(32))) _Float16 bsh[512];   // 32k x 16t fragment order
    const int tid  = threadIdx.x;
    const int lane = tid & 31;
    const int wave = tid >> 5;
    const int nlin = lane & 15;
    const int h    = lane >> 4;
    const int b    = blockIdx.x >> 5;          // Tc/16 == 32 t-tiles per batch
    const int t0   = (blockIdx.x & 31) * 16;
    const int XC   = (MODE == 0) ? 128 : CK;   // real channel count of X

    const float* Xb = X + (size_t)b * XC * Tc;
    const int kt  = tid >> 2;                  // staged k_local 0..31
    const int tl4 = (tid & 3) * 4;             // staged t_local base

    v8f acc[4];
#pragma unroll
    for (int mi = 0; mi < 4; ++mi) acc[mi] = (v8f){0.f,0.f,0.f,0.f,0.f,0.f,0.f,0.f};

    for (int kc = 0; kc < CK; kc += 32) {
        __syncthreads();                       // previous chunk fully consumed
        // ---- cooperative stage of B chunk into fragment order ----
        const int hb = inv_h_(kt);
        const int eb = inv_e_(kt);
        if (MODE == 0) {
            const int kp = kc + kt;            // k' = r*128 + c
            const int r  = kp >> 7;
            const int c  = kp & 127;
#pragma unroll
            for (int q = 0; q < 4; ++q) {
                const int tl   = tl4 + q;
                const int tcol = t0 + tl + r - 1;
                const int tcl  = imin_(imax_(tcol, 0), Tc - 1);
                const float mk = (tcol >= 0 && tcol < Tc) ? 1.0f : 0.0f;
                bsh[((tl + 16 * hb) << 4) + eb] =
                    (_Float16)(Xb[(size_t)c * Tc + tcl] * mk);
            }
        } else {
            const float4 xv = *(const float4*)&Xb[(size_t)(kc + kt) * Tc + t0 + tl4];
            bsh[((tl4 + 0 + 16 * hb) << 4) + eb] = (_Float16)xv.x;
            bsh[((tl4 + 1 + 16 * hb) << 4) + eb] = (_Float16)xv.y;
            bsh[((tl4 + 2 + 16 * hb) << 4) + eb] = (_Float16)xv.z;
            bsh[((tl4 + 3 + 16 * hb) << 4) + eb] = (_Float16)xv.w;
        }
        __syncthreads();
        // ---- one B fragment feeds 4 WMMAs per wave (16 per block) ----
        const v16h bf = *(const v16h*)&bsh[lane << 4];
#pragma unroll
        for (int mi = 0; mi < 4; ++mi) {
            const int mt = wave + mi * 4;
            const v16h af = *(const v16h*)&Wpk[
                (size_t)(((mt * (CK / 32) + (kc >> 5)) * 32 + lane) << 4)];
            acc[mi] = __builtin_amdgcn_wmma_f32_16x16x32_f16(
                false, af, false, bf, (short)0, acc[mi], false, false);
        }
    }
    // ---- epilogue (C/D layout: lane n = lane&15, M = 8*h + rr) ----
    const int t = t0 + nlin;
#pragma unroll
    for (int mi = 0; mi < 4; ++mi) {
        const int mt = wave + mi * 4;
#pragma unroll
        for (int rr = 0; rr < 8; ++rr) {
            const int m = mt * 16 + h * 8 + rr;
            float v = acc[mi][rr];
            if (MODE == 0) {
                v = siluf_(v + bias[m]);
                v = bnf_(v, bng[m], bnb[m], bnm[m], bnv[m]);
                out[(size_t)b * Mc * Tc + (size_t)m * Tc + t] = v;
            } else if (MODE == 1) {
                v = bnf_(v + bias[m], bng[m], bnb[m], bnm[m], bnv[m]);
                out[(size_t)b * Mc * Tc + (size_t)m * Tc + t] = siluf_(v);
            } else if (MODE == 2) {
                v = bnf_(v + bias[m], bng[m], bnb[m], bnm[m], bnv[m]);
                out[(size_t)b * Mc * Tc + (size_t)m * Tc + t] =
                    v + resid[(size_t)b * Mc * Tc + (size_t)m * Tc + t];
            } else {
                v = fmaxf(v + bias[m], 0.0f);
                out[(size_t)b * Tc * Mc + (size_t)t * Mc + m] = NYQ * sigmoidf_(v);
            }
        }
    }
}

// depthwise conv3 (pad 1, groups=C) -> bn2 -> silu, branch-free boundaries
__global__ void dw_k(const float* __restrict__ Hin, float* __restrict__ Dout,
                     const float* __restrict__ dww, const float* __restrict__ dwb,
                     const float* __restrict__ g, const float* __restrict__ bb,
                     const float* __restrict__ m_, const float* __restrict__ v_) {
    const int idx = blockIdx.x * blockDim.x + threadIdx.x;
    if (idx >= Bc * Mc * Tc) return;
    const int t = idx & (Tc - 1);
    const int c = (idx >> 9) & (Mc - 1);
    const float* row = Hin + (size_t)(idx - t);
    const float xm = row[imax_(t - 1, 0)] * ((t > 0) ? 1.0f : 0.0f);
    const float xp = row[imin_(t + 1, Tc - 1)] * ((t < Tc - 1) ? 1.0f : 0.0f);
    const float acc = dwb[c] + dww[c * 3] * xm + dww[c * 3 + 1] * row[t] + dww[c * 3 + 2] * xp;
    Dout[idx] = siluf_(bnf_(acc, g[c], bb[c], m_[c], v_[c]));
}

// SE: mean over T
__global__ void se_mean_k(const float* __restrict__ D, float* __restrict__ s) {
    const int idx = blockIdx.x * blockDim.x + threadIdx.x;  // b*256+c
    if (idx >= Bc * Mc) return;
    const float4* row = (const float4*)(D + (size_t)idx * Tc);
    float acc = 0.f;
    for (int t = 0; t < Tc / 4; ++t) {
        const float4 v = row[t];
        acc += v.x + v.y + v.z + v.w;
    }
    s[idx] = acc * (1.0f / Tc);
}

// SE MLP: s1 = silu(s@W1+b1) [B,64]; s2 = sigmoid(s1@W2+b2) [B,256]
__global__ void se_mlp_k(const float* __restrict__ s, float* __restrict__ s2,
                         const float* __restrict__ W1, const float* __restrict__ b1,
                         const float* __restrict__ W2, const float* __restrict__ b2) {
    __shared__ float s1s[Bc * 64];
    const int tid = threadIdx.x;                 // 1024 threads, 1 block
    if (tid < Bc * 64) {
        const int b = tid >> 6, r = tid & 63;
        float acc = b1[r];
        for (int c = 0; c < Mc; ++c) acc += s[b * Mc + c] * W1[c * 64 + r];
        s1s[tid] = siluf_(acc);
    }
    __syncthreads();
    const int b = tid >> 8, c = tid & 255;
    float acc = b2[c];
    for (int r = 0; r < 64; ++r) acc += s1s[b * 64 + r] * W2[r * Mc + c];
    s2[tid] = sigmoidf_(acc);
}

__global__ void se_scale_k(float* __restrict__ D, const float* __restrict__ s2) {
    const int idx = blockIdx.x * blockDim.x + threadIdx.x;
    if (idx >= Bc * Mc * Tc) return;
    D[idx] *= s2[idx >> 9];
}

// amps[b,t,d] = exp_sigmoid(a[b,d,t])
__global__ void amps_k(const float* __restrict__ A, float* __restrict__ AMPS) {
    const int idx = blockIdx.x * blockDim.x + threadIdx.x;   // b*T*256 + t*256 + d
    if (idx >= Bc * Tc * Mc) return;
    const int d = idx & 255, t = (idx >> 8) & (Tc - 1), b = idx >> 17;
    AMPS[idx] = expsig_(A[(size_t)b * Mc * Tc + (size_t)d * Tc + t]);
}

// Phase prefix table: P[b,j,k] = sum of lin-interp freq env over samples in
// frame-intervals < j (pre 2pi/SR scaling).
__global__ void phase_tab_k(const float* __restrict__ FRQ, float* __restrict__ PT) {
    const int tid = blockIdx.x * blockDim.x + threadIdx.x;   // b*256+k
    if (tid >= Bc * Mc) return;
    const int b = tid >> 8, k = tid & 255;
    const float s = 511.0f / 65535.0f;
    float prefix = 0.f;
    int mj = 0;
    for (int j = 0; j < Tc; ++j) {
        const int mjn = (j < Tc - 1) ? ((j + 1) * 65535 + 510) / 511 : NS;
        const int cnt = mjn - mj;
        const int j1  = (j < Tc - 1) ? j + 1 : j;
        const float F0 = FRQ[(size_t)b * Tc * Mc + (size_t)j  * Mc + k];
        const float F1 = FRQ[(size_t)b * Tc * Mc + (size_t)j1 * Mc + k];
        PT[(size_t)b * Tc * Mc + (size_t)j * Mc + k] = prefix;
        const float sumw = s * 0.5f * (float)(mj + mjn - 1) * (float)cnt
                         - (float)j * (float)cnt;
        prefix += (float)cnt * F0 + (F1 - F0) * sumw;
        mj = mjn;
    }
}

// Oscillator bank: audio[b,n] = sum_k sin(phase)*amp, closed-form phase.
__global__ void osc_k(const float* __restrict__ FRQ, const float* __restrict__ AMPS,
                      const float* __restrict__ PT, float* __restrict__ out) {
    const int idx = blockIdx.x * blockDim.x + threadIdx.x;   // b*NS + n
    if (idx >= Bc * NS) return;
    const int n = idx & (NS - 1), b = idx >> 16;
    const float s    = 511.0f / 65535.0f;
    const float c2pi = 6.283185307179586f / SRF;
    const int num = n * 511;                   // exact: pos = n*511/65535
    const int j   = num / 65535;
    const int i1  = (j < Tc - 1) ? j + 1 : j;
    const float w = (float)(num - j * 65535) * (1.0f / 65535.0f);
    const int mj  = (j * 65535 + 510) / 511;   // first sample of interval j
    const float q1 = (float)(n - mj + 1);
    const float q2 = s * 0.5f * (float)(mj + n) * q1 - (float)j * q1;

    const float4* F0p = (const float4*)(FRQ  + (size_t)b * Tc * Mc + (size_t)j  * Mc);
    const float4* F1p = (const float4*)(FRQ  + (size_t)b * Tc * Mc + (size_t)i1 * Mc);
    const float4* A0p = (const float4*)(AMPS + (size_t)b * Tc * Mc + (size_t)j  * Mc);
    const float4* A1p = (const float4*)(AMPS + (size_t)b * Tc * Mc + (size_t)i1 * Mc);
    const float4* Pp  = (const float4*)(PT   + (size_t)b * Tc * Mc + (size_t)j  * Mc);
    __builtin_prefetch(F1p, 0, 0);   // -> global_prefetch_b8
    float acc = 0.f;
    for (int k4 = 0; k4 < Mc / 4; ++k4) {
        const float4 f0 = F0p[k4], f1 = F1p[k4];
        const float4 a0 = A0p[k4], a1 = A1p[k4];
        const float4 pp = Pp[k4];
#pragma unroll
        for (int e = 0; e < 4; ++e) {
            const float F0 = ((const float*)&f0)[e];
            const float dF = ((const float*)&f1)[e] - F0;
            const float ph = c2pi * (((const float*)&pp)[e] + q1 * F0 + dF * q2);
            const float fr = F0 + dF * w;
            const float A0 = ((const float*)&a0)[e];
            const float am = A0 + (((const float*)&a1)[e] - A0) * w;
            acc += __sinf(ph) * ((fr < NYQ) ? am : 0.0f);
        }
    }
    out[idx] = acc;
}

// ---------------------------------------------------------------------------
// Host side.  Input order assumption: d_in[0] = z, then params leaves in
// sorted-nested-dict (jax tree_flatten) order:
//  1:l1.b 2:l1.w 3..6:net1.bn{b,g,m,v} 7:net1.cb 8:net1.cw
//  9..20:net1.mb.bn{1,2,3}{b,g,m,v} 21:dw_b 22:dw_w 23:pw1_b 24:pw1_w
//  25:pw2_b 26:pw2_w  27..30:net2.bn 31:cb 32:cw 33..44:net2 bn1-3
//  45:dw_b 46:dw_w 47:pw1_b 48:pw1_w 49:pw2_b 50:pw2_w
//  51:se_b1 52:se_b2 53:se_w1 54:se_w2
// ---------------------------------------------------------------------------
extern "C" void kernel_launch(void* const* d_in, const int* in_sizes, int n_in,
                              void* d_out, int out_size, void* d_ws, size_t ws_size,
                              hipStream_t stream) {
    (void)in_sizes; (void)n_in; (void)out_size; (void)ws_size;
    auto F = [&](int i) { return (const float*)d_in[i]; };
    const float* z = F(0);
    const float* l1b = F(1); const float* l1w = F(2);

    float* ws = (float*)d_ws;
    const size_t CH = (size_t)Bc * Mc * Tc;      // 524288
    float* X1   = ws;             // bn(silu(conv)) net1  (mb residual input)
    float* H    = ws + 1 * CH;    // pw1 output
    float* D    = ws + 2 * CH;    // dw output
    float* AFIN = ws + 3 * CH;    // net1 final [B,256,T]
    float* X2   = ws + 4 * CH;    // net2 residual input
    float* FFIN = ws + 5 * CH;    // net2 final [B,256,T]
    float* AMPS = ws + 6 * CH;    // [B,T,256]
    float* FRQS = ws + 7 * CH;    // [B,T,256]
    float* PT   = ws + 8 * CH;    // phase prefix [B,512,256]
    float* SE   = ws + 9 * CH;    // s[1024], s2[1024]
    float* s_   = SE;
    float* s2_  = SE + 1024;
    // packed f16 weights (fragment order)
    _Float16* wp   = (_Float16*)(ws + 9 * CH + 4096);
    _Float16* W1c  = wp;                       // 256*384
    _Float16* W1p1 = W1c  + 256 * 384;         // 256*256
    _Float16* W1p2 = W1p1 + 256 * 256;
    _Float16* W2c  = W1p2 + 256 * 256;         // 256*384
    _Float16* W2p1 = W2c  + 256 * 384;
    _Float16* W2p2 = W2p1 + 256 * 256;
    _Float16* Wl1  = W2p2 + 256 * 256;

    const dim3 gG(Bc * (Tc / 16)), bG(128);
    const dim3 gE((Bc * Mc * Tc + 255) / 256), bE(256);
    const dim3 gPc(Mc * 384 / 256), gPp(Mc * 256 / 256);

    // ---- pack all weights to f16 fragment order ----
    hipLaunchKernelGGL((pack_k<0, 384>), gPc, bE, 0, stream, F(8),  W1c);
    hipLaunchKernelGGL((pack_k<1, 256>), gPp, bE, 0, stream, F(24), W1p1);
    hipLaunchKernelGGL((pack_k<2, 256>), gPp, bE, 0, stream, F(26), W1p2);
    hipLaunchKernelGGL((pack_k<0, 384>), gPc, bE, 0, stream, F(32), W2c);
    hipLaunchKernelGGL((pack_k<1, 256>), gPp, bE, 0, stream, F(48), W2p1);
    hipLaunchKernelGGL((pack_k<2, 256>), gPp, bE, 0, stream, F(50), W2p2);
    hipLaunchKernelGGL((pack_k<3, 256>), gPp, bE, 0, stream, l1w,   Wl1);

    // ---- net1 ----
    hipLaunchKernelGGL((gemm_k<0, 384>), gG, bG, 0, stream,
                       z, W1c, X1, F(7), F(4), F(3), F(5), F(6), nullptr);
    hipLaunchKernelGGL((gemm_k<1, 256>), gG, bG, 0, stream,
                       X1, W1p1, H, F(23), F(10), F(9), F(11), F(12), nullptr);
    hipLaunchKernelGGL(dw_k, gE, bE, 0, stream,
                       H, D, F(22), F(21), F(14), F(13), F(15), F(16));
    hipLaunchKernelGGL((gemm_k<2, 256>), gG, bG, 0, stream,
                       D, W1p2, AFIN, F(25), F(18), F(17), F(19), F(20), X1);

    // ---- net2 ----
    hipLaunchKernelGGL((gemm_k<0, 384>), gG, bG, 0, stream,
                       z, W2c, X2, F(31), F(28), F(27), F(29), F(30), nullptr);
    hipLaunchKernelGGL((gemm_k<1, 256>), gG, bG, 0, stream,
                       X2, W2p1, H, F(47), F(34), F(33), F(35), F(36), nullptr);
    hipLaunchKernelGGL(dw_k, gE, bE, 0, stream,
                       H, D, F(46), F(45), F(38), F(37), F(39), F(40));
    hipLaunchKernelGGL(se_mean_k, dim3(4), bE, 0, stream, D, s_);
    hipLaunchKernelGGL(se_mlp_k, dim3(1), dim3(1024), 0, stream,
                       s_, s2_, F(53), F(51), F(54), F(52));
    hipLaunchKernelGGL(se_scale_k, gE, bE, 0, stream, D, s2_);
    hipLaunchKernelGGL((gemm_k<2, 256>), gG, bG, 0, stream,
                       D, W2p2, FFIN, F(49), F(42), F(41), F(43), F(44), X2);

    // ---- l1 + controls ----
    hipLaunchKernelGGL((gemm_k<3, 256>), gG, bG, 0, stream,
                       FFIN, Wl1, FRQS, l1b, nullptr, nullptr, nullptr, nullptr, nullptr);
    hipLaunchKernelGGL(amps_k, gE, bE, 0, stream, AFIN, AMPS);

    // ---- oscillator bank ----
    hipLaunchKernelGGL(phase_tab_k, dim3(4), bE, 0, stream, FRQS, PT);
    hipLaunchKernelGGL(osc_k, dim3(Bc * NS / 256), bE, 0, stream,
                       FRQS, AMPS, PT, (float*)d_out);
}
